// EnhancedMambaBlock_52871047414041
// MI455X (gfx1250) — matrix operand order repaired and checked by reading, not compiled, not run on hardware
//
#include <hip/hip_runtime.h>
#include <hip/hip_bf16.h>

#define BB     2
#define CCH    128
#define HDIM   56
#define WDIM   56
#define HWX    3136      // 56*56
#define LL     3136
#define DI     256       // d_inner
#define NS     16        // d_state
#define NCHUNK 98
#define CLEN   32        // NCHUNK*CLEN == LL
#define MROWS  (BB*LL)   // 6272

typedef __attribute__((ext_vector_type(16))) __bf16 v16bf;
typedef __attribute__((ext_vector_type(8)))  __bf16 v8bf;
typedef __attribute__((ext_vector_type(8)))  float  v8f;

struct bfpair { v8bf lo, hi; };

__device__ __forceinline__ __bf16 f2bf(float f) {
  unsigned u = __builtin_bit_cast(unsigned, f);
  u = (u + 0x7FFFu + ((u >> 16) & 1u)) >> 16;
  unsigned short s = (unsigned short)u;
  return __builtin_bit_cast(__bf16, s);
}
__device__ __forceinline__ float sigm(float v) { return 1.0f / (1.0f + __expf(-v)); }
__device__ __forceinline__ float silu(float v) { return v * sigm(v); }

// Streaming f32 -> bf16 cast; 8-element runs past nvalid8 are zero-filled
// (used to pad weight matrices to a multiple of 64 rows).
__global__ __launch_bounds__(256) void cast_bf16_kernel(
    const float* __restrict__ src, __bf16* __restrict__ dst, int n8, int nvalid8)
{
  int i = blockIdx.x * 256 + threadIdx.x;
  if (i >= n8) return;
  int base = i * 8;
  v8bf o;
  if (i < nvalid8) {
#pragma unroll
    for (int j = 0; j < 8; ++j) o[j] = f2bf(src[base + j]);
  } else {
#pragma unroll
    for (int j = 0; j < 8; ++j) o[j] = f2bf(0.0f);
  }
  *reinterpret_cast<v8bf*>(dst + base) = o;
}

// ---------------------------------------------------------------------------
// GEMM: C[M,N] = A[M,K] @ W[N,K]^T (+bias). Pre-cast bf16 operands (W padded
// to 64-row multiple), f32 accumulate. One 16x64 C strip per wave32:
// 4 accumulators, A-frag reused 4x, unconditional aligned b128 loads.
// ---------------------------------------------------------------------------
__global__ __launch_bounds__(128) void gemm_bf16_wmma(
    const __bf16* __restrict__ Ab, const __bf16* __restrict__ Wb,
    const float* __restrict__ bias, float* __restrict__ Cg,
    int M, int N, int K, int hasBias)
{
  const int tilesN = (N + 63) >> 6;
  const int tilesM = M >> 4;                    // M is multiple of 16
  int wv = blockIdx.x * 4 + (threadIdx.x >> 5);
  if (wv >= tilesM * tilesN) return;
  int tm = wv / tilesN, tn = wv % tilesN;
  int lane = threadIdx.x & 31;
  int half = lane >> 4, l16 = lane & 15;
  const __bf16* Arow = Ab + (size_t)(tm * 16 + l16) * K;
  const __bf16* Wr[4];
#pragma unroll
  for (int j = 0; j < 4; ++j)                   // rows < Npad: always valid
    Wr[j] = Wb + (size_t)(tn * 64 + 16 * j + l16) * K;

  v8f acc[4] = {};
  for (int k0 = 0; k0 < K; k0 += 32) {
    // A 16x32 bf16 frag: lane half -> K runs [half*8,+8) and [16+half*8,+8)
    v8bf alo = *reinterpret_cast<const v8bf*>(Arow + k0 + half * 8);
    v8bf ahi = *reinterpret_cast<const v8bf*>(Arow + k0 + 16 + half * 8);
    bfpair ap{alo, ahi};
    v16bf af = __builtin_bit_cast(v16bf, ap);
#pragma unroll
    for (int j = 0; j < 4; ++j) {
      // B 32x16 frag: column n, K runs [half*8,+8) and [16+half*8,+8)
      v8bf blo = *reinterpret_cast<const v8bf*>(Wr[j] + k0 + half * 8);
      v8bf bhi = *reinterpret_cast<const v8bf*>(Wr[j] + k0 + 16 + half * 8);
      bfpair bp{blo, bhi};
      v16bf bfr = __builtin_bit_cast(v16bf, bp);
      acc[j] = __builtin_amdgcn_wmma_f32_16x16x32_bf16(false, af, false, bfr,
                                                       (short)0, acc[j], false, false);
    }
  }
#pragma unroll
  for (int j = 0; j < 4; ++j) {
    int nOut = tn * 64 + 16 * j + l16;
    if (nOut < N) {
      float bv = hasBias ? bias[nOut] : 0.0f;
#pragma unroll
      for (int r = 0; r < 8; ++r) {
        int mo = tm * 16 + half * 8 + r;        // C: VGPR r -> row half*8+r
        Cg[(size_t)mo * N + nOut] = acc[j][r] + bv;
      }
    }
  }
}

// ---------------------------------------------------------------------------
// LayerNorm over channels; emits both row-major (h) and col-major (v) seqs.
// ---------------------------------------------------------------------------
__global__ __launch_bounds__(128) void ln_kernel(
    const float* __restrict__ x,
    const float* __restrict__ hw, const float* __restrict__ hb,
    const float* __restrict__ vw, const float* __restrict__ vb,
    float* __restrict__ xh, float* __restrict__ xv)
{
  int pos = blockIdx.x;                 // b*HW + h*W + w
  int b = pos / HWX, rem = pos % HWX;
  int hh = rem / WDIM, ww = rem % WDIM;
  int c = threadIdx.x;
  float val = x[(((size_t)b * CCH + c) * HDIM + hh) * WDIM + ww];
  __shared__ float s1[128], s2[128];
  s1[c] = val; s2[c] = val * val;
  __syncthreads();
  for (int s = 64; s > 0; s >>= 1) {
    if (c < s) { s1[c] += s1[c + s]; s2[c] += s2[c + s]; }
    __syncthreads();
  }
  float mean = s1[0] * (1.0f / 128.0f);
  float var  = s2[0] * (1.0f / 128.0f) - mean * mean;
  float xn = (val - mean) * rsqrtf(var + 1e-5f);
  int lh = hh * WDIM + ww, lv = ww * HDIM + hh;
  xh[((size_t)b * HWX + lh) * CCH + c] = xn * hw[c] + hb[c];
  xv[((size_t)b * HWX + lv) * CCH + c] = xn * vw[c] + vb[c];
}

// Causal depthwise conv (width 4) along L on xi = xz[:, :, 0:256], then SiLU.
__global__ __launch_bounds__(256) void convsilu_kernel(
    const float* __restrict__ xz, const float* __restrict__ cw,
    const float* __restrict__ cb, float* __restrict__ xs)
{
  int idx = blockIdx.x * 256 + threadIdx.x;
  if (idx >= BB * LL * DI) return;
  int d = idx & 255, bl = idx >> 8;
  int l = bl % LL, b = bl / LL;
  float acc = cb[d];
#pragma unroll
  for (int j = 0; j < 4; ++j) {
    int ls = l - 3 + j;
    if (ls >= 0) acc += cw[d * 4 + j] * xz[(size_t)(b * LL + ls) * 512 + d];
  }
  xs[idx] = silu(acc);
}

// dt = softplus(dbl[:, :8] @ dt_w^T + dt_b)
__global__ __launch_bounds__(256) void dt_kernel(
    const float* __restrict__ dbl, const float* __restrict__ dtw,
    const float* __restrict__ dtb, float* __restrict__ dt)
{
  int idx = blockIdx.x * 256 + threadIdx.x;
  if (idx >= BB * LL * DI) return;
  int d = idx & 255, row = idx >> 8;
  const float* r = dbl + (size_t)row * 40;
  float acc = dtb[d];
#pragma unroll
  for (int j = 0; j < 8; ++j) acc += r[j] * dtw[d * 8 + j];
  dt[idx] = (acc > 20.0f) ? acc : __logf(1.0f + __expf(acc));
}

// -------- Chunked parallel scan: h_l = exp(dt*A)*h + dt*xs*B,  y = <h,C> ----
__global__ __launch_bounds__(256) void scan_chunk_kernel(
    const float* __restrict__ dt, const float* __restrict__ xs,
    const float* __restrict__ dbl, const float* __restrict__ A_log,
    float* __restrict__ P, float* __restrict__ U)
{
  int chunk = blockIdx.x, b = blockIdx.y, d = threadIdx.x;
  __shared__ float sBC[CLEN][32];           // [step][0..15]=B, [16..31]=C
  int l0 = chunk * CLEN;
  for (int i = threadIdx.x; i < CLEN * 32; i += 256) {
    int s = i >> 5, j = i & 31;
    sBC[s][j] = dbl[(size_t)(b * LL + l0 + s) * 40 + 8 + j];
  }
  __syncthreads();
  float a[NS], p[NS], u[NS];
#pragma unroll
  for (int n = 0; n < NS; ++n) { a[n] = -__expf(A_log[d * NS + n]); p[n] = 1.0f; u[n] = 0.0f; }
  for (int s = 0; s < CLEN; ++s) {
    size_t bl = (size_t)(b * LL + l0 + s);
    float dtv = dt[bl * DI + d];
    float dx = dtv * xs[bl * DI + d];
#pragma unroll
    for (int n = 0; n < NS; ++n) {
      float da = __expf(dtv * a[n]);
      u[n] = da * u[n] + dx * sBC[s][n];
      p[n] *= da;
    }
  }
  size_t base = (((size_t)b * NCHUNK + chunk) * DI + d) * NS;
#pragma unroll
  for (int n = 0; n < NS; ++n) { P[base + n] = p[n]; U[base + n] = u[n]; }
}

__global__ __launch_bounds__(256) void scan_prop_kernel(
    const float* __restrict__ P, const float* __restrict__ U, float* __restrict__ Hs)
{
  int b = blockIdx.x, d = threadIdx.x;
  float h[NS];
#pragma unroll
  for (int n = 0; n < NS; ++n) h[n] = 0.0f;
  for (int c = 0; c < NCHUNK; ++c) {
    size_t base = (((size_t)b * NCHUNK + c) * DI + d) * NS;
#pragma unroll
    for (int n = 0; n < NS; ++n) {
      Hs[base + n] = h[n];
      h[n] = P[base + n] * h[n] + U[base + n];
    }
  }
}

__global__ __launch_bounds__(256) void scan_final_kernel(
    const float* __restrict__ dt, const float* __restrict__ xs,
    const float* __restrict__ dbl, const float* __restrict__ A_log,
    const float* __restrict__ Dp, const float* __restrict__ xz,
    const float* __restrict__ Hs, float* __restrict__ y)
{
  int chunk = blockIdx.x, b = blockIdx.y, d = threadIdx.x;
  __shared__ float sBC[CLEN][32];
  int l0 = chunk * CLEN;
  for (int i = threadIdx.x; i < CLEN * 32; i += 256) {
    int s = i >> 5, j = i & 31;
    sBC[s][j] = dbl[(size_t)(b * LL + l0 + s) * 40 + 8 + j];
  }
  __syncthreads();
  float a[NS], h[NS];
  size_t hb = (((size_t)b * NCHUNK + chunk) * DI + d) * NS;
#pragma unroll
  for (int n = 0; n < NS; ++n) { a[n] = -__expf(A_log[d * NS + n]); h[n] = Hs[hb + n]; }
  float Dv = Dp[d];
  for (int s = 0; s < CLEN; ++s) {
    size_t bl = (size_t)(b * LL + l0 + s);
    float dtv = dt[bl * DI + d];
    float xv  = xs[bl * DI + d];
    float dx = dtv * xv;
    float acc = 0.0f;
#pragma unroll
    for (int n = 0; n < NS; ++n) {
      float da = __expf(dtv * a[n]);
      h[n] = da * h[n] + dx * sBC[s][n];
      acc += h[n] * sBC[s][16 + n];
    }
    float z = xz[bl * 512 + DI + d];
    y[bl * DI + d] = (acc + xv * Dv) * silu(z);
  }
}

// Depthwise 3x3 + exact GELU, emitted transposed to (B, HW, C) for the GEMM.
__global__ __launch_bounds__(256) void dwgelu_kernel(
    const float* __restrict__ x, const float* __restrict__ w,
    const float* __restrict__ bb, float* __restrict__ t)
{
  int idx = blockIdx.x * 256 + threadIdx.x;
  if (idx >= BB * HWX * CCH) return;
  int c = idx & 127, bl = idx >> 7;
  int l = bl % HWX, b = bl / HWX;
  int hh = l / WDIM, ww = l % WDIM;
  float acc = bb[c];
#pragma unroll
  for (int dh = -1; dh <= 1; ++dh)
#pragma unroll
    for (int dw = -1; dw <= 1; ++dw) {
      int hs = hh + dh, wsd = ww + dw;
      if (hs >= 0 && hs < HDIM && wsd >= 0 && wsd < WDIM)
        acc += w[c * 9 + (dh + 1) * 3 + (dw + 1)] *
               x[(((size_t)b * CCH + c) * HDIM + hs) * WDIM + wsd];
    }
  t[idx] = 0.5f * acc * (1.0f + erff(acc * 0.70710678118654752f));
}

__global__ __launch_bounds__(256) void fuse_kernel(
    const float* __restrict__ oh, const float* __restrict__ ov,
    const float* __restrict__ lf, float* __restrict__ fused)
{
  int idx = blockIdx.x * 256 + threadIdx.x;
  if (idx >= BB * CCH * HWX) return;
  int ww = idx % WDIM;
  int hh = (idx / WDIM) % HDIM;
  int c  = (idx / HWX) % CCH;
  int b  = idx / (CCH * HWX);
  int lh = hh * WDIM + ww, lv = ww * HDIM + hh;
  fused[idx] = oh[((size_t)b * HWX + lh) * CCH + c] +
               ov[((size_t)b * HWX + lv) * CCH + c] +
               lf[((size_t)b * HWX + lh) * CCH + c];
}

__global__ __launch_bounds__(256) void mean_kernel(
    const float* __restrict__ fused, float* __restrict__ mean)
{
  int bc = blockIdx.x;
  const float* p = fused + (size_t)bc * HWX;
  float s = 0.0f;
  for (int i = threadIdx.x; i < HWX; i += 256) s += p[i];
  __shared__ float sm[256];
  sm[threadIdx.x] = s;
  __syncthreads();
  for (int st = 128; st > 0; st >>= 1) {
    if (threadIdx.x < st) sm[threadIdx.x] += sm[threadIdx.x + st];
    __syncthreads();
  }
  if (threadIdx.x == 0) mean[bc] = sm[0] * (1.0f / (float)HWX);
}

__global__ __launch_bounds__(256) void ca_kernel(
    const float* __restrict__ mean, const float* __restrict__ fc1,
    const float* __restrict__ fc2, float* __restrict__ scale)
{
  __shared__ float sh[BB * 32];
  int t = threadIdx.x;
  if (t < BB * 32) {
    int b = t >> 5, r = t & 31;
    float acc = 0.0f;
    for (int c = 0; c < CCH; ++c) acc += mean[b * CCH + c] * fc1[r * CCH + c];
    sh[b * 32 + r] = fmaxf(acc, 0.0f);
  }
  __syncthreads();
  {
    int b = t >> 7, o = t & 127;
    float acc = 0.0f;
#pragma unroll
    for (int r = 0; r < 32; ++r) acc += sh[b * 32 + r] * fc2[o * 32 + r];
    scale[b * CCH + o] = sigm(acc);
  }
}

__global__ __launch_bounds__(256) void final_kernel(
    const float* __restrict__ fused, const float* __restrict__ scale,
    const float* __restrict__ x, float* __restrict__ out)
{
  int idx = blockIdx.x * 256 + threadIdx.x;
  if (idx >= BB * CCH * HWX) return;
  int bc = idx / HWX;
  out[idx] = fused[idx] * scale[bc] + x[idx];
}

// ---------------------------------------------------------------------------
static inline void launch_gemm(const float* A, const float* Wt, const float* bias,
                               float* C, int M, int N, int K,
                               __bf16* Abf, __bf16* Wbf, hipStream_t s) {
  int Npad = ((N + 63) >> 6) << 6;
  int na8 = (M * K) / 8;
  int nw8 = (Npad * K) / 8, nwv8 = (N * K) / 8;
  cast_bf16_kernel<<<(na8 + 255) / 256, 256, 0, s>>>(A, Abf, na8, na8);
  cast_bf16_kernel<<<(nw8 + 255) / 256, 256, 0, s>>>(Wt, Wbf, nw8, nwv8);
  int tiles = (M >> 4) * (Npad >> 6);
  gemm_bf16_wmma<<<(tiles + 3) >> 2, 128, 0, s>>>(Abf, Wbf, bias, C, M, N, K,
                                                  bias ? 1 : 0);
}

extern "C" void kernel_launch(void* const* d_in, const int* in_sizes, int n_in,
                              void* d_out, int out_size, void* d_ws, size_t ws_size,
                              hipStream_t stream) {
  (void)in_sizes; (void)n_in; (void)out_size; (void)ws_size;
  const float* x        = (const float*)d_in[0];
  const float* norm_h_w = (const float*)d_in[1];
  const float* norm_h_b = (const float*)d_in[2];
  const float* norm_v_w = (const float*)d_in[3];
  const float* norm_v_b = (const float*)d_in[4];
  const float* dw_w     = (const float*)d_in[5];
  const float* dw_b     = (const float*)d_in[6];
  const float* pw_w     = (const float*)d_in[7];
  const float* pw_b     = (const float*)d_in[8];
  // h: 9..17, v: 18..26  (in_w, conv_w, conv_b, xproj_w, dt_w, dt_b, A_log, D, out_w)
  const float* ca_fc1_w = (const float*)d_in[27];
  const float* ca_fc2_w = (const float*)d_in[28];

  float* ws = (float*)d_ws;
  size_t off = 0;
  auto alloc = [&](size_t n) { float* p = ws + off; off += n; return p; };
  const size_t BLC = (size_t)BB * HWX * CCH;        // 802816
  float* xh   = alloc(BLC);
  float* xv   = alloc(BLC);
  float* oh   = alloc(BLC);
  float* ov   = alloc(BLC);
  float* lf   = alloc(BLC);
  float* xz   = alloc((size_t)BB * LL * 512);
  float* xs   = alloc((size_t)BB * LL * DI);
  float* dtb  = alloc((size_t)BB * LL * DI);
  float* yb   = alloc((size_t)BB * LL * DI);
  float* dbl  = alloc((size_t)BB * LL * 40);
  float* Pb   = alloc((size_t)BB * NCHUNK * DI * NS);
  float* Ub   = alloc((size_t)BB * NCHUNK * DI * NS);
  float* Hsb  = alloc((size_t)BB * NCHUNK * DI * NS);
  __bf16* Abf = (__bf16*)alloc((size_t)MROWS * DI / 2);   // 6272*256 bf16
  __bf16* Wbf = (__bf16*)alloc((size_t)512 * 128 / 2);    // max Npad*K bf16
  float* mnb  = alloc(BB * CCH);
  float* scb  = alloc(BB * CCH);
  float* tbuf  = xz;   // alias: xz dead once both mamba dirs complete
  float* fused = xs;   // alias: xs dead after second scan_final

  // 1) LayerNorm -> both sequence orders
  ln_kernel<<<BB * HWX, 128, 0, stream>>>(x, norm_h_w, norm_h_b, norm_v_w, norm_v_b, xh, xv);

  // 2) Mamba (h then v), sharing scratch
  auto run_mamba = [&](const float* xin, const float* const* wp, float* obuf) {
    const float* in_w   = wp[0]; const float* conv_w = wp[1];
    const float* conv_b = wp[2]; const float* xproj_w = wp[3];
    const float* dt_w   = wp[4]; const float* dt_bias = wp[5];
    const float* A_log  = wp[6]; const float* Dp = wp[7];
    const float* out_w  = wp[8];
    launch_gemm(xin, in_w, nullptr, xz, MROWS, 512, 128, Abf, Wbf, stream);    // in_proj
    convsilu_kernel<<<(BB * LL * DI + 255) / 256, 256, 0, stream>>>(xz, conv_w, conv_b, xs);
    launch_gemm(xs, xproj_w, nullptr, dbl, MROWS, 40, 256, Abf, Wbf, stream);  // x_proj
    dt_kernel<<<(BB * LL * DI + 255) / 256, 256, 0, stream>>>(dbl, dt_w, dt_bias, dtb);
    scan_chunk_kernel<<<dim3(NCHUNK, BB), 256, 0, stream>>>(dtb, xs, dbl, A_log, Pb, Ub);
    scan_prop_kernel<<<BB, 256, 0, stream>>>(Pb, Ub, Hsb);
    scan_final_kernel<<<dim3(NCHUNK, BB), 256, 0, stream>>>(dtb, xs, dbl, A_log,
                                                            Dp, xz, Hsb, yb);
    launch_gemm(yb, out_w, nullptr, obuf, MROWS, 128, 256, Abf, Wbf, stream);  // out_proj
  };
  const float* hW[9], *vW[9];
  for (int i = 0; i < 9; ++i) { hW[i] = (const float*)d_in[9 + i]; vW[i] = (const float*)d_in[18 + i]; }
  run_mamba(xh, hW, oh);
  run_mamba(xv, vW, ov);

  // 3) Local feature: dw3x3 + GELU (transposed) then pointwise GEMM (+bias)
  dwgelu_kernel<<<(BB * HWX * CCH + 255) / 256, 256, 0, stream>>>(x, dw_w, dw_b, tbuf);
  launch_gemm(tbuf, pw_w, pw_b, lf, MROWS, 128, 128, Abf, Wbf, stream);

  // 4) Fuse + channel attention + residual
  fuse_kernel<<<(BB * CCH * HWX + 255) / 256, 256, 0, stream>>>(oh, ov, lf, fused);
  mean_kernel<<<BB * CCH, 256, 0, stream>>>(fused, mnb);
  ca_kernel<<<1, 256, 0, stream>>>(mnb, ca_fc1_w, ca_fc2_w, scb);
  final_kernel<<<(BB * CCH * HWX + 255) / 256, 256, 0, stream>>>(fused, scb, x, (float*)d_out);
}
